// AttentionRNNSLU_24635932410303
// MI455X (gfx1250) — matrix operand order repaired and checked by reading, not compiled, not run on hardware
//
#include <hip/hip_runtime.h>
#include <hip/hip_bf16.h>

// ---------------------------------------------------------------------------
// AttentionRNNSLU forward for MI455X (gfx1250, wave32, WMMA).
// All matmuls go through v_wmma_f32_16x16x32_bf16 (bf16 in, f32 accum) with
// register double-buffering over K (overlap global_load_b128 with WMMA) and
// global_prefetch_b8 for upcoming weight tiles.
// Recurrence handled by stream-ordered launches (graph-capture safe).
// Workspace requirement: ~192 MB.
// ---------------------------------------------------------------------------

constexpr int B_    = 256;   // batch
constexpr int T_    = 128;   // seq len
constexpr int E_    = 768;   // bert emb
constexpr int HE_   = 256;   // encoder hidden per direction
constexpr int H_    = 512;   // decoder hidden (= 2*HE)
constexpr int SLOT_ = 120;
constexpr int SLOTP_= 128;   // SLOT padded to WMMA tile
constexpr int INTENT_ = 64;
constexpr int EMB_  = 40;
constexpr int DIN_  = 1064;  // EMB + 2H
constexpr int KENC_ = 1024;  // E + HE  (fused encoder GEMM K)
constexpr int KDEC_ = 1600;  // DIN + H = 1576 padded to multiple of 64
constexpr int NENC_ = 1024;  // 4*HE
constexpr int NDEC_ = 2048;  // 4*H

#define CDIV(a,b) (((a)+(b)-1)/(b))

typedef __attribute__((ext_vector_type(16))) __bf16          v16bf;
typedef __attribute__((ext_vector_type(8)))  float           v8f;
typedef __attribute__((ext_vector_type(8)))  unsigned short  v8us;

__device__ __forceinline__ unsigned short f2bf(float f) {
  unsigned int u = __builtin_bit_cast(unsigned int, f);
  u += 0x7fffu + ((u >> 16) & 1u);          // round-to-nearest-even
  return (unsigned short)(u >> 16);
}
__device__ __forceinline__ float sigm(float x) { return 1.0f / (1.0f + expf(-x)); }

__device__ __forceinline__ v16bf cmb(v8us lo, v8us hi) {
  return __builtin_bit_cast(v16bf,
      __builtin_shufflevector(lo, hi, 0,1,2,3,4,5,6,7,8,9,10,11,12,13,14,15));
}

// One K-tile (K=32) worth of fragments: A 16x32, B 32x64 (4 n-tiles).
struct KTile { v8us alo, ahi; v8us blo[4], bhi[4]; };

// Double-buffered WMMA inner loop. K must be a multiple of 64.
// arowh = A row base (+half*8); wrow[nt] = W row base (+half*8).
__device__ __forceinline__ void gemm_body(const unsigned short* __restrict__ arowh,
                                          const unsigned short* const* wrow,
                                          int K, v8f (&acc)[4]) {
  KTile t0, t1;
  auto ld = [&](KTile& t, int k0) {
    t.alo = *(const v8us*)(arowh + k0);
    t.ahi = *(const v8us*)(arowh + k0 + 16);
#pragma unroll
    for (int nt = 0; nt < 4; ++nt) {
      t.blo[nt] = *(const v8us*)(wrow[nt] + k0);
      t.bhi[nt] = *(const v8us*)(wrow[nt] + k0 + 16);
    }
  };
  auto mm = [&](KTile& t) {
    v16bf a = cmb(t.alo, t.ahi);
#pragma unroll
    for (int nt = 0; nt < 4; ++nt)
      acc[nt] = __builtin_amdgcn_wmma_f32_16x16x32_bf16(
          false, a, false, cmb(t.blo[nt], t.bhi[nt]), (short)0, acc[nt], false, false);
  };
  ld(t0, 0);
  for (int k0 = 0; k0 < K; k0 += 64) {
    ld(t1, k0 + 32);                      // loads for tile t+1 in flight ...
    if (k0 + 320 < K) {                   // and prefetch a few tiles ahead
      __builtin_prefetch(arowh + k0 + 320, 0, 1);
#pragma unroll
      for (int nt = 0; nt < 4; ++nt) __builtin_prefetch(wrow[nt] + k0 + 320, 0, 1);
    }
    mm(t0);                               // ... while WMMAs consume tile t
    if (k0 + 64 < K) ld(t0, k0 + 64);
    mm(t1);
  }
}

// ---------------------------------------------------------------------------
// WMMA GEMM:  C[M,N](f32) = A[M,K](bf16, row-major, ld=lda)
//                         @ W[N,K](bf16, row-major, ld=ldw)^T  + bias[N]
// Block = 256 threads = 8 waves; block tile 128x64; wave tile 16x64.
// grid = (M/128, N/64); K multiple of 64.
// Fragment layouts per CDNA5 ISA 7.12.2 (wave32).
// ---------------------------------------------------------------------------
__global__ __launch_bounds__(256)
void gemm_bf16_wmma(const unsigned short* __restrict__ A, int lda,
                    const unsigned short* __restrict__ W, int ldw,
                    float* __restrict__ C, int ldc,
                    const float* __restrict__ bias, int K)
{
  const int wave = threadIdx.x >> 5;
  const int lane = threadIdx.x & 31;
  const int half = lane >> 4;
  const int l16  = lane & 15;
  const int m0 = blockIdx.x * 128 + wave * 16;
  const int n0 = blockIdx.y * 64;

  const unsigned short* arowh = A + (size_t)(m0 + l16) * lda + half * 8;
  const unsigned short* wrow[4];
#pragma unroll
  for (int nt = 0; nt < 4; ++nt)
    wrow[nt] = W + (size_t)(n0 + nt * 16 + l16) * ldw + half * 8;

  v8f acc[4] = {};
  gemm_body(arowh, wrow, K, acc);

  const int row = m0 + half * 8;
#pragma unroll
  for (int nt = 0; nt < 4; ++nt) {
    const int col = n0 + nt * 16 + l16;
    const float bv = bias ? bias[col] : 0.0f;
#pragma unroll
    for (int r = 0; r < 8; ++r)
      C[(size_t)(row + r) * ldc + col] = acc[nt][r] + bv;
  }
}

// Fused two-problem variant (encoder fwd+bwd in one launch, blockIdx.z selects).
__global__ __launch_bounds__(256)
void gemm_bf16_wmma_x2(const unsigned short* __restrict__ A0,
                       const unsigned short* __restrict__ A1, int lda,
                       const unsigned short* __restrict__ W0,
                       const unsigned short* __restrict__ W1, int ldw,
                       float* __restrict__ C0, float* __restrict__ C1, int ldc,
                       const float* __restrict__ bias0,
                       const float* __restrict__ bias1, int K)
{
  const unsigned short* A = blockIdx.z ? A1 : A0;
  const unsigned short* W = blockIdx.z ? W1 : W0;
  float* C = blockIdx.z ? C1 : C0;
  const float* bias = blockIdx.z ? bias1 : bias0;

  const int wave = threadIdx.x >> 5;
  const int lane = threadIdx.x & 31;
  const int half = lane >> 4;
  const int l16  = lane & 15;
  const int m0 = blockIdx.x * 128 + wave * 16;
  const int n0 = blockIdx.y * 64;

  const unsigned short* arowh = A + (size_t)(m0 + l16) * lda + half * 8;
  const unsigned short* wrow[4];
#pragma unroll
  for (int nt = 0; nt < 4; ++nt)
    wrow[nt] = W + (size_t)(n0 + nt * 16 + l16) * ldw + half * 8;

  v8f acc[4] = {};
  gemm_body(arowh, wrow, K, acc);

  const int row = m0 + half * 8;
#pragma unroll
  for (int nt = 0; nt < 4; ++nt) {
    const int col = n0 + nt * 16 + l16;
    const float bv = bias ? bias[col] : 0.0f;
#pragma unroll
    for (int r = 0; r < 8; ++r)
      C[(size_t)(row + r) * ldc + col] = acc[nt][r] + bv;
  }
}

// ---------------------------------------------------------------------------
// Utility kernels
// ---------------------------------------------------------------------------
__global__ void zero_us(unsigned short* p, size_t n) {
  size_t i = (size_t)blockIdx.x * blockDim.x + threadIdx.x;
  size_t s = (size_t)gridDim.x * blockDim.x;
  for (; i < n; i += s) p[i] = 0;
}
__global__ void zero_f(float* p, size_t n) {
  size_t i = (size_t)blockIdx.x * blockDim.x + threadIdx.x;
  size_t s = (size_t)gridDim.x * blockDim.x;
  for (; i < n; i += s) p[i] = 0.0f;
}
// dst[r*dst_ld + c] = bf16(src[r*src_ld + c]) for r<rows, c<cols
__global__ void cvt_pack(const float* __restrict__ src, unsigned short* __restrict__ dst,
                         int rows, int cols, int src_ld, int dst_ld) {
  int i = blockIdx.x * blockDim.x + threadIdx.x;
  if (i >= rows * cols) return;
  int r = i / cols, c = i % cols;
  dst[(size_t)r * dst_ld + c] = f2bf(src[(size_t)r * src_ld + c]);
}
__global__ void add_bias(const float* a, const float* b, float* o, int n) {
  int i = blockIdx.x * blockDim.x + threadIdx.x;
  if (i < n) o[i] = a[i] + b[i];
}
__global__ void pad_bias(const float* s, float* d, int n, int ntot) {
  int i = blockIdx.x * blockDim.x + threadIdx.x;
  if (i < ntot) d[i] = (i < n) ? s[i] : 0.0f;
}
__global__ void init_emb(const float* __restrict__ tab, unsigned short* __restrict__ e) {
  int i = blockIdx.x * blockDim.x + threadIdx.x;
  if (i < B_ * EMB_) e[i] = f2bf(tab[SLOT_ * EMB_ + (i % EMB_)]);
}

// ---------------------------------------------------------------------------
// Encoder step kernels
// ---------------------------------------------------------------------------
// X_f[b] = [ bf16(seq[b,t]) , h_fwd[b] ] ; X_b[b] = [ bf16(seq[b,T-1-t]) , h_bwd[b] ]
__global__ void enc_build_x(const float* __restrict__ seq,
                            const unsigned short* __restrict__ hfb, // [2][B][HE] bf16
                            unsigned short* __restrict__ Xf,
                            unsigned short* __restrict__ Xb, int t) {
  int i = blockIdx.x * blockDim.x + threadIdx.x;
  if (i >= B_ * KENC_) return;
  int b = i / KENC_, k = i % KENC_;
  unsigned short vf, vb;
  if (k < E_) {
    vf = f2bf(seq[((size_t)b * T_ + t) * E_ + k]);
    vb = f2bf(seq[((size_t)b * T_ + (T_ - 1 - t)) * E_ + k]);
  } else {
    vf = hfb[b * HE_ + (k - E_)];
    vb = hfb[(size_t)B_ * HE_ + b * HE_ + (k - E_)];
  }
  Xf[i] = vf;
  Xb[i] = vb;
}

// gates[2][B][4HE] -> h,c ; writes enc f32 + bf16, h bf16 for next step
__global__ void enc_cell(const float* __restrict__ gates,
                         float* __restrict__ c,            // [2][B][HE]
                         unsigned short* __restrict__ hbf, // [2][B][HE]
                         float* __restrict__ enc,          // [B][T][H]
                         unsigned short* __restrict__ encbf, int t) {
  int i = blockIdx.x * blockDim.x + threadIdx.x;
  if (i >= 2 * B_ * HE_) return;
  int dir = i / (B_ * HE_);
  int r = i % (B_ * HE_);
  int b = r / HE_, j = r % HE_;
  const float* g = gates + ((size_t)dir * B_ + b) * (4 * HE_);
  float ig = sigm(g[j]);
  float fg = sigm(g[HE_ + j]);
  float gg = tanhf(g[2 * HE_ + j]);
  float og = sigm(g[3 * HE_ + j]);
  float cn = fg * c[i] + ig * gg;
  c[i] = cn;
  float hn = og * tanhf(cn);
  hbf[i] = f2bf(hn);
  int tout = (dir == 0) ? t : (T_ - 1 - t);
  size_t eo = ((size_t)b * T_ + tout) * H_ + dir * HE_ + j;
  enc[eo] = hn;
  encbf[eo] = f2bf(hn);
}

// last_hidden -> initial decoder ctx
__global__ __launch_bounds__(128)
void init_ctx_last_hidden(const int* __restrict__ nwp, const float* __restrict__ enc,
                          float* __restrict__ ctx, unsigned short* __restrict__ ctx_bf) {
  int b = blockIdx.x, tid = threadIdx.x;
  __shared__ int red[128];
  red[tid] = (nwp[b * T_ + tid] != 0) ? 1 : 0;
  __syncthreads();
  for (int s = 64; s > 0; s >>= 1) { if (tid < s) red[tid] += red[tid + s]; __syncthreads(); }
  int cnt = min(red[0], T_ - 1);
  for (int k = tid; k < H_; k += 128) {
    float v = enc[((size_t)b * T_ + cnt) * H_ + k];
    ctx[(size_t)b * H_ + k] = v;
    ctx_bf[(size_t)b * H_ + k] = f2bf(v);
  }
}

// ---------------------------------------------------------------------------
// Decoder step kernels
// ---------------------------------------------------------------------------
// X[b] = [emb(40) | ctx(512) | enc_t(512) | h(512) | pad(24)=0]; also stash old
// ctx into hc[:,512:1024] for the slot-score GEMM.
__global__ void dec_build_x(const unsigned short* __restrict__ emb_bf,
                            const unsigned short* __restrict__ ctx_bf,
                            const unsigned short* __restrict__ enc_bf,
                            unsigned short* __restrict__ hc, // [B][1024]; h in [:,0:512]
                            unsigned short* __restrict__ X, int t) {
  int i = blockIdx.x * blockDim.x + threadIdx.x;
  if (i >= B_ * KDEC_) return;
  int b = i / KDEC_, k = i % KDEC_;
  unsigned short v;
  if (k < EMB_)              v = emb_bf[b * EMB_ + k];
  else if (k < EMB_ + H_)    v = ctx_bf[b * H_ + (k - EMB_)];
  else if (k < DIN_)         v = enc_bf[((size_t)b * T_ + t) * H_ + (k - EMB_ - H_)];
  else if (k < DIN_ + H_)    v = hc[b * 1024 + (k - DIN_)];
  else                       v = 0;
  X[(size_t)b * KDEC_ + k] = v;
  if (k < H_) hc[b * 1024 + 512 + k] = ctx_bf[b * H_ + k]; // old ctx for score
}

__global__ void dec_cell(const float* __restrict__ gates, // [B][4H]
                         float* __restrict__ c, float* __restrict__ h,
                         unsigned short* __restrict__ hc) { // h bf16 -> hc[:,0:512]
  int i = blockIdx.x * blockDim.x + threadIdx.x;
  if (i >= B_ * H_) return;
  int b = i / H_, j = i % H_;
  const float* g = gates + (size_t)b * (4 * H_);
  float ig = sigm(g[j]);
  float fg = sigm(g[H_ + j]);
  float gg = tanhf(g[2 * H_ + j]);
  float og = sigm(g[3 * H_ + j]);
  float cn = fg * c[i] + ig * gg;
  c[i] = cn;
  float hn = og * tanhf(cn);
  h[i] = hn;
  hc[b * 1024 + j] = f2bf(hn);
}

// log_softmax over 120 slots, argmax -> next embedding (bf16)
__global__ __launch_bounds__(128)
void slot_head(const float* __restrict__ score, const float* __restrict__ emb_tab,
               float* __restrict__ out_slot, unsigned short* __restrict__ emb_bf, int t) {
  int b = blockIdx.x, j = threadIdx.x;
  __shared__ float sm[128];
  __shared__ int   si[128];
  float v = (j < SLOT_) ? score[b * SLOTP_ + j] : -INFINITY;
  sm[j] = v; si[j] = j;
  __syncthreads();
  for (int s = 64; s > 0; s >>= 1) {
    if (j < s) {
      float o = sm[j + s]; int oi = si[j + s];
      if (o > sm[j] || (o == sm[j] && oi < si[j])) { sm[j] = o; si[j] = oi; }
    }
    __syncthreads();
  }
  float mx = sm[0]; int amax = si[0];
  __syncthreads();
  sm[j] = (j < SLOT_) ? expf(v - mx) : 0.0f;
  __syncthreads();
  for (int s = 64; s > 0; s >>= 1) { if (j < s) sm[j] += sm[j + s]; __syncthreads(); }
  float lse = mx + logf(sm[0]);
  if (j < SLOT_) out_slot[((size_t)b * T_ + t) * SLOT_ + j] = v - lse;
  if (j < EMB_)  emb_bf[b * EMB_ + j] = f2bf(emb_tab[amax * EMB_ + j]);
}

// masked softmax attention: ctx = softmax(energies . h, mask) @ enc
__global__ __launch_bounds__(128)
void attention_step(const float* __restrict__ energies, const float* __restrict__ hdec,
                    const float* __restrict__ enc, const unsigned char* __restrict__ mask,
                    float* __restrict__ ctx, unsigned short* __restrict__ ctx_bf) {
  int b = blockIdx.x, tid = threadIdx.x;
  __shared__ float red[128];
  __shared__ float sal[T_];
  const float* ept = energies + ((size_t)b * T_ + tid) * H_;
  const float* hb  = hdec + (size_t)b * H_;
  float e = 0.0f;
  for (int k = 0; k < H_; ++k) e += ept[k] * hb[k];
  if (mask[b * T_ + tid]) e = -1e12f;
  red[tid] = e;
  __syncthreads();
  for (int s = 64; s > 0; s >>= 1) { if (tid < s) red[tid] = fmaxf(red[tid], red[tid + s]); __syncthreads(); }
  float mx = red[0];
  __syncthreads();
  float ex = expf(e - mx);
  sal[tid] = ex; red[tid] = ex;
  __syncthreads();
  for (int s = 64; s > 0; s >>= 1) { if (tid < s) red[tid] += red[tid + s]; __syncthreads(); }
  float inv = 1.0f / red[0];
  __syncthreads();
  for (int k = tid; k < H_; k += 128) {
    float acc = 0.0f;
    for (int tt = 0; tt < T_; ++tt) acc += sal[tt] * enc[((size_t)b * T_ + tt) * H_ + k];
    acc *= inv;
    ctx[(size_t)b * H_ + k] = acc;
    ctx_bf[(size_t)b * H_ + k] = f2bf(acc);
  }
}

// [h0 | ctx'0] bf16 for the intent head
__global__ void build_hc_int(const unsigned short* __restrict__ hc,
                             const unsigned short* __restrict__ ctx_bf,
                             unsigned short* __restrict__ hci) {
  int i = blockIdx.x * blockDim.x + threadIdx.x;
  if (i >= B_ * 1024) return;
  int b = i / 1024, k = i % 1024;
  hci[i] = (k < H_) ? hc[b * 1024 + k] : ctx_bf[b * H_ + (k - H_)];
}

// ---------------------------------------------------------------------------
// Host orchestration
// ---------------------------------------------------------------------------
extern "C" void kernel_launch(void* const* d_in, const int* in_sizes, int n_in,
                              void* d_out, int out_size, void* d_ws, size_t ws_size,
                              hipStream_t stream) {
  (void)in_sizes; (void)n_in; (void)out_size; (void)ws_size;

  const float*         seq     = (const float*)d_in[0];
  const int*           nwp     = (const int*)d_in[1];
  const unsigned char* mask    = (const unsigned char*)d_in[2];
  const float*         emb_tab = (const float*)d_in[3];
  const float* Wf_ih = (const float*)d_in[4];  const float* Wf_hh = (const float*)d_in[5];
  const float* bf_ih = (const float*)d_in[6];  const float* bf_hh = (const float*)d_in[7];
  const float* Wb_ih = (const float*)d_in[8];  const float* Wb_hh = (const float*)d_in[9];
  const float* bb_ih = (const float*)d_in[10]; const float* bb_hh = (const float*)d_in[11];
  const float* Wd_ih = (const float*)d_in[12]; const float* Wd_hh = (const float*)d_in[13];
  const float* bd_ih = (const float*)d_in[14]; const float* bd_hh = (const float*)d_in[15];
  const float* attnW = (const float*)d_in[16]; const float* attnb = (const float*)d_in[17];
  const float* slotW = (const float*)d_in[18]; const float* slotb = (const float*)d_in[19];
  const float* intW  = (const float*)d_in[20]; const float* intb  = (const float*)d_in[21];

  float* out_slot = (float*)d_out;
  float* out_int  = out_slot + (size_t)B_ * T_ * SLOT_;

  char* p = (char*)d_ws;
  auto alloc = [&](size_t bytes) -> char* {
    char* r = p; p += (bytes + 255) & ~(size_t)255; return r;
  };

  // bf16 buffers
  unsigned short* W_enc_f  = (unsigned short*)alloc((size_t)NENC_ * KENC_ * 2);
  unsigned short* W_enc_b  = (unsigned short*)alloc((size_t)NENC_ * KENC_ * 2);
  unsigned short* W_dec    = (unsigned short*)alloc((size_t)NDEC_ * KDEC_ * 2);
  unsigned short* W_attn   = (unsigned short*)alloc((size_t)H_ * H_ * 2);
  unsigned short* W_slot   = (unsigned short*)alloc((size_t)SLOTP_ * 1024 * 2);
  unsigned short* W_int    = (unsigned short*)alloc((size_t)INTENT_ * 1024 * 2);
  unsigned short* enc_bf   = (unsigned short*)alloc((size_t)B_ * T_ * H_ * 2);
  unsigned short* Xf       = (unsigned short*)alloc((size_t)B_ * KENC_ * 2);
  unsigned short* Xb       = (unsigned short*)alloc((size_t)B_ * KENC_ * 2);
  unsigned short* Xd       = (unsigned short*)alloc((size_t)B_ * KDEC_ * 2);
  unsigned short* hc_cat   = (unsigned short*)alloc((size_t)B_ * 1024 * 2);
  unsigned short* hc_int   = (unsigned short*)alloc((size_t)B_ * 1024 * 2);
  unsigned short* ctx_bf   = (unsigned short*)alloc((size_t)B_ * H_ * 2);
  unsigned short* emb_bf   = (unsigned short*)alloc((size_t)B_ * EMB_ * 2);
  unsigned short* h_enc_bf = (unsigned short*)alloc((size_t)2 * B_ * HE_ * 2);
  // f32 buffers
  float* enc_f    = (float*)alloc((size_t)B_ * T_ * H_ * 4);
  float* energies = (float*)alloc((size_t)B_ * T_ * H_ * 4);
  float* gates_e  = (float*)alloc((size_t)2 * B_ * NENC_ * 4);
  float* gates_d  = (float*)alloc((size_t)B_ * NDEC_ * 4);
  float* score    = (float*)alloc((size_t)B_ * SLOTP_ * 4);
  float* c_enc    = (float*)alloc((size_t)2 * B_ * HE_ * 4);
  float* h_dec    = (float*)alloc((size_t)B_ * H_ * 4);
  float* c_dec    = (float*)alloc((size_t)B_ * H_ * 4);
  float* ctx_f    = (float*)alloc((size_t)B_ * H_ * 4);
  float* bias_ef  = (float*)alloc(NENC_ * 4);
  float* bias_eb  = (float*)alloc(NENC_ * 4);
  float* bias_d   = (float*)alloc(NDEC_ * 4);
  float* bias_s   = (float*)alloc(SLOTP_ * 4);

  // ---- weight packing (deterministic, every call) ----
  zero_us<<<2048, 256, 0, stream>>>(W_dec,  (size_t)NDEC_ * KDEC_);
  zero_us<<<512,  256, 0, stream>>>(W_slot, (size_t)SLOTP_ * 1024);
  cvt_pack<<<CDIV(NENC_*E_,256),  256, 0, stream>>>(Wf_ih, W_enc_f,       NENC_, E_,   E_,   KENC_);
  cvt_pack<<<CDIV(NENC_*HE_,256), 256, 0, stream>>>(Wf_hh, W_enc_f + E_,  NENC_, HE_,  HE_,  KENC_);
  cvt_pack<<<CDIV(NENC_*E_,256),  256, 0, stream>>>(Wb_ih, W_enc_b,       NENC_, E_,   E_,   KENC_);
  cvt_pack<<<CDIV(NENC_*HE_,256), 256, 0, stream>>>(Wb_hh, W_enc_b + E_,  NENC_, HE_,  HE_,  KENC_);
  cvt_pack<<<CDIV(NDEC_*DIN_,256),256, 0, stream>>>(Wd_ih, W_dec,         NDEC_, DIN_, DIN_, KDEC_);
  cvt_pack<<<CDIV(NDEC_*H_,256),  256, 0, stream>>>(Wd_hh, W_dec + DIN_,  NDEC_, H_,   H_,   KDEC_);
  cvt_pack<<<CDIV(H_*H_,256),     256, 0, stream>>>(attnW, W_attn,        H_,    H_,   H_,   H_);
  cvt_pack<<<CDIV(SLOT_*1024,256),256, 0, stream>>>(slotW, W_slot,        SLOT_, 1024, 1024, 1024);
  cvt_pack<<<CDIV(INTENT_*1024,256),256,0,stream>>>(intW,  W_int,         INTENT_,1024,1024, 1024);
  add_bias<<<CDIV(NENC_,256), 256, 0, stream>>>(bf_ih, bf_hh, bias_ef, NENC_);
  add_bias<<<CDIV(NENC_,256), 256, 0, stream>>>(bb_ih, bb_hh, bias_eb, NENC_);
  add_bias<<<CDIV(NDEC_,256), 256, 0, stream>>>(bd_ih, bd_hh, bias_d,  NDEC_);
  pad_bias<<<1, 128, 0, stream>>>(slotb, bias_s, SLOT_, SLOTP_);

  // ---- state init ----
  zero_us<<<256, 256, 0, stream>>>(h_enc_bf, (size_t)2 * B_ * HE_);
  zero_f <<<256, 256, 0, stream>>>(c_enc,    (size_t)2 * B_ * HE_);
  zero_f <<<256, 256, 0, stream>>>(h_dec,    (size_t)B_ * H_);
  zero_f <<<256, 256, 0, stream>>>(c_dec,    (size_t)B_ * H_);
  zero_us<<<256, 256, 0, stream>>>(hc_cat,   (size_t)B_ * 1024);
  init_emb<<<CDIV(B_*EMB_,256), 256, 0, stream>>>(emb_tab, emb_bf);

  // ---- encoder: bidirectional LSTM, fused [x,h] GEMM per step ----
  for (int t = 0; t < T_; ++t) {
    enc_build_x<<<CDIV(B_*KENC_,256), 256, 0, stream>>>(seq, h_enc_bf, Xf, Xb, t);
    gemm_bf16_wmma_x2<<<dim3(B_/128, NENC_/64, 2), 256, 0, stream>>>(
        Xf, Xb, KENC_, W_enc_f, W_enc_b, KENC_,
        gates_e, gates_e + (size_t)B_ * NENC_, NENC_, bias_ef, bias_eb, KENC_);
    enc_cell<<<CDIV(2*B_*HE_,256), 256, 0, stream>>>(gates_e, c_enc, h_enc_bf, enc_f, enc_bf, t);
  }

  // ---- last hidden -> initial ctx; precompute attention energies ----
  init_ctx_last_hidden<<<B_, 128, 0, stream>>>(nwp, enc_f, ctx_f, ctx_bf);
  gemm_bf16_wmma<<<dim3((B_*T_)/128, H_/64), 256, 0, stream>>>(
      enc_bf, H_, W_attn, H_, energies, H_, attnb, H_);

  // ---- decoder ----
  for (int t = 0; t < T_; ++t) {
    dec_build_x<<<CDIV(B_*KDEC_,256), 256, 0, stream>>>(emb_bf, ctx_bf, enc_bf, hc_cat, Xd, t);
    gemm_bf16_wmma<<<dim3(B_/128, NDEC_/64), 256, 0, stream>>>(
        Xd, KDEC_, W_dec, KDEC_, gates_d, NDEC_, bias_d, KDEC_);
    dec_cell<<<CDIV(B_*H_,256), 256, 0, stream>>>(gates_d, c_dec, h_dec, hc_cat);
    gemm_bf16_wmma<<<dim3(B_/128, SLOTP_/64), 256, 0, stream>>>(
        hc_cat, 1024, W_slot, 1024, score, SLOTP_, bias_s, 1024);
    slot_head<<<B_, 128, 0, stream>>>(score, emb_tab, out_slot, emb_bf, t);
    attention_step<<<B_, 128, 0, stream>>>(energies, h_dec, enc_f, mask, ctx_f, ctx_bf);
    if (t == 0) {
      build_hc_int<<<CDIV(B_*1024,256), 256, 0, stream>>>(hc_cat, ctx_bf, hc_int);
      gemm_bf16_wmma<<<dim3(B_/128, INTENT_/64), 256, 0, stream>>>(
          hc_int, 1024, W_int, 1024, out_int, INTENT_, intb, 1024);
    }
  }
}
